// QuerySelection_68092411510991
// MI455X (gfx1250) — matrix operand order repaired and checked by reading, not compiled, not run on hardware
//
#include <hip/hip_runtime.h>
#include <hip/hip_bf16.h>
#include <math.h>

// ---------------------------------------------------------------------------
// QuerySelection fused pipeline for gfx1250 (MI455X), bf16 WMMA everywhere.
// B=32, S=8400, H=256, C=80, NUM_QUERIES=300.
// ---------------------------------------------------------------------------

typedef __bf16 bf16;
typedef __attribute__((ext_vector_type(16))) __bf16 v16bf;
typedef __attribute__((ext_vector_type(8)))  __bf16 v8bf;
typedef __attribute__((ext_vector_type(8)))  float  v8f;

#define S_TOT 8400
#define BATCH 32
#define HDIM 256
#define NCLS 80
#define NQ 300
#define NEG_BIG (-3.402823e38f)

// One 16x16 (MxN) WMMA tile over the full K=256 reduction.
// A: LDS, row-major [rows][256] bf16.
//   ISA 16-bit A 16x32 layout: lanes 0-15 = rows, VGPR0-3 hold K 0..7 (koff 0)
//   and VGPR4-7 hold K 16..23; lanes 16-31 same rows, koff = 8.
// Wt: global, bf16 [256][ldb] = transposed weights, and the B fragment is 16
//   scalar loads per lane, ldb apart (lane%16 = N, K in the same order as A).
__device__ __forceinline__ v8f wmma_k256(const bf16* __restrict__ A, int lda,
                                         const bf16* __restrict__ Wt, int ldb,
                                         int m_base, int n0, int lane, v8f acc) {
  const int arow = m_base + (lane & 15);
  const int koff = (lane >> 4) * 8;
#pragma unroll
  for (int kk = 0; kk < 256; kk += 32) {
    const bf16* pa = A + arow * lda + kk + koff;
    v8bf lo = *(const v8bf*)pa;
    v8bf hi = *(const v8bf*)(pa + 16);
    v16bf af;
#pragma unroll
    for (int i = 0; i < 8; ++i) { af[i] = lo[i]; af[i + 8] = hi[i]; }
    v16bf bfm;
#pragma unroll
    for (int i = 0; i < 8; ++i) {
      bfm[i] = Wt[(size_t)(kk + koff + i) * ldb + n0 + (lane & 15)];
      bfm[i + 8] = Wt[(size_t)(kk + 16 + koff + i) * ldb + n0 + (lane & 15)];
    }
    acc = __builtin_amdgcn_wmma_f32_16x16x32_bf16(false, af, false, bfm,
                                                  (short)0, acc, false, false);
  }
  return acc;
}

// ---------------------------------------------------------------------------
// Kernel 1: transpose + convert all weights fp32 [N][K] -> bf16 [K][Npad].
// Total bf16 elements: 3*65536 + 20480 + 4096 = 221184 = 864 blocks * 256.
// ---------------------------------------------------------------------------
__global__ __launch_bounds__(256) void qs_prep_weights(
    const float* __restrict__ pw, const float* __restrict__ sw,
    const float* __restrict__ w1, const float* __restrict__ w2,
    const float* __restrict__ w3,
    bf16* __restrict__ projT, bf16* __restrict__ scoreT,
    bf16* __restrict__ b1T, bf16* __restrict__ b2T, bf16* __restrict__ b3T) {
  int idx = blockIdx.x * 256 + threadIdx.x;
  if (idx < 65536) {
    int k = idx >> 8, n = idx & 255;
    projT[idx] = (bf16)pw[n * 256 + k];
  } else if (idx < 131072) {
    int j = idx - 65536; int k = j >> 8, n = j & 255;
    b1T[j] = (bf16)w1[n * 256 + k];
  } else if (idx < 196608) {
    int j = idx - 131072; int k = j >> 8, n = j & 255;
    b2T[j] = (bf16)w2[n * 256 + k];
  } else if (idx < 217088) {
    int j = idx - 196608; int k = j / 80, n = j % 80;
    scoreT[j] = (bf16)sw[n * 256 + k];
  } else if (idx < 221184) {
    int j = idx - 217088; int k = j >> 4, n = j & 15;   // pad N: 4 -> 16
    b3T[j] = (n < 4) ? (bf16)w3[n * 256 + k] : (bf16)0.f;
  }
}

// ---------------------------------------------------------------------------
// Kernel 2: anchors [S][4] + validity mask [S] (spatial shapes are static).
// ---------------------------------------------------------------------------
__global__ __launch_bounds__(256) void qs_anchors(float* __restrict__ anchors,
                                                  float* __restrict__ mask) {
  int s = blockIdx.x * 256 + threadIdx.x;
  if (s >= S_TOT) return;
  int lvl, w, base;
  if (s < 6400)      { lvl = 0; w = 80; base = 0;    }
  else if (s < 8000) { lvl = 1; w = 40; base = 6400; }
  else               { lvl = 2; w = 20; base = 8000; }
  int h = w;
  int sl = s - base;
  int gy = sl / w, gx = sl % w;
  float a[4];
  a[0] = (gx + 0.5f) / (float)w;
  a[1] = (gy + 0.5f) / (float)h;
  a[2] = 0.05f * (float)(1 << lvl);
  a[3] = a[2];
  bool valid = true;
#pragma unroll
  for (int i = 0; i < 4; ++i) valid = valid && (a[i] > 0.01f) && (a[i] < 0.99f);
#pragma unroll
  for (int i = 0; i < 4; ++i) {
    float ac = fminf(fmaxf(a[i], 1e-4f), 1.0f - 1e-4f);
    anchors[s * 4 + i] = logf(ac / (1.0f - ac));
  }
  mask[s] = valid ? 1.0f : 0.0f;
}

// ---------------------------------------------------------------------------
// Kernel 3: main pass. Per block: 32 tokens of one batch.
// mask -> write masked_memory; bf16 tile; proj WMMA + bias; LayerNorm;
// score WMMA + bias; per-row max -> maxscore[B,S].
// ---------------------------------------------------------------------------
#define P2_TILES 263   // ceil(8400/32)

__global__ __launch_bounds__(256) void qs_phase2(
    const float* __restrict__ memory, const float* __restrict__ mask,
    const bf16* __restrict__ projT, const bf16* __restrict__ scoreT,
    const float* __restrict__ proj_b, const float* __restrict__ ln_g,
    const float* __restrict__ ln_b, const float* __restrict__ score_b,
    float* __restrict__ mm_out, float* __restrict__ maxscore) {
  __shared__ __align__(64) bf16 sA[32 * 256];   // A tile, reused as encoded
  __shared__ __align__(64) float sC[32 * 256];  // f32 GEMM results
  __shared__ float sMu[32], sRstd[32];

  const int tid = threadIdx.x;
  const int wave = tid >> 5, lane = tid & 31;
  const int b  = blockIdx.x / P2_TILES;
  const int m0 = (blockIdx.x % P2_TILES) * 32;

  // load + mask + emit masked_memory + bf16 convert
  for (int idx = tid; idx < 32 * 256; idx += 256) {
    int r = idx >> 8, c = idx & 255;
    int row = m0 + r;
    float v = 0.f;
    if (row < S_TOT) {
      size_t g = ((size_t)b * S_TOT + row) * 256 + c;
      v = memory[g] * mask[row];
      mm_out[g] = v;
    }
    sA[idx] = (bf16)v;
  }
  __syncthreads();

  // proj GEMM: 2 (M) x 16 (N) tiles over 8 waves, 4 tiles/wave
  for (int t = wave; t < 32; t += 8) {
    int mt = t >> 4, nt = t & 15;
    v8f acc = {};
    acc = wmma_k256(sA, 256, projT, 256, mt * 16, nt * 16, lane, acc);
    int cb = nt * 16 + (lane & 15);
    int rb = mt * 16 + ((lane >> 4) << 3);
    float bias = proj_b[cb];
#pragma unroll
    for (int i = 0; i < 8; ++i) sC[(rb + i) * 256 + cb] = acc[i] + bias;
  }
  __syncthreads();

  // LayerNorm stats
  if (tid < 32) {
    float s1 = 0.f, s2 = 0.f;
    for (int c = 0; c < 256; ++c) { float v = sC[tid * 256 + c]; s1 += v; s2 += v * v; }
    float mu = s1 * (1.f / 256.f);
    float var = s2 * (1.f / 256.f) - mu * mu;
    sMu[tid] = mu; sRstd[tid] = rsqrtf(var + 1e-5f);
  }
  __syncthreads();

  // normalize -> encoded bf16 (overwrites sA; all A reads done)
  for (int idx = tid; idx < 32 * 256; idx += 256) {
    int r = idx >> 8, c = idx & 255;
    float v = (sC[idx] - sMu[r]) * sRstd[r] * ln_g[c] + ln_b[c];
    sA[idx] = (bf16)v;
  }
  __syncthreads();

  // score GEMM: 2 x 5 tiles over 8 waves
  for (int t = wave; t < 10; t += 8) {
    int mt = t / 5, nt = t % 5;
    v8f acc = {};
    acc = wmma_k256(sA, 256, scoreT, 80, mt * 16, nt * 16, lane, acc);
    int cb = nt * 16 + (lane & 15);
    int rb = mt * 16 + ((lane >> 4) << 3);
    float bias = score_b[cb];
#pragma unroll
    for (int i = 0; i < 8; ++i) sC[(rb + i) * 80 + cb] = acc[i] + bias;
  }
  __syncthreads();

  // per-row max over 80 classes
  if (tid < 32) {
    int row = m0 + tid;
    if (row < S_TOT) {
      float m = sC[tid * 80];
      for (int c = 1; c < 80; ++c) m = fmaxf(m, sC[tid * 80 + c]);
      maxscore[(size_t)b * S_TOT + row] = m;
    }
  }
}

// ---------------------------------------------------------------------------
// Kernel 4: per-batch top-300 (descending score, lowest index on ties,
// matching jax.lax.top_k). Scores staged in LDS; wave32 shuffle reduction.
// ---------------------------------------------------------------------------
__global__ __launch_bounds__(256) void qs_topk(const float* __restrict__ maxscore,
                                               int* __restrict__ topk) {
  __shared__ float sms[S_TOT];
  __shared__ float rv[8];
  __shared__ int   ri[8];
  const int b = blockIdx.x, tid = threadIdx.x;
  const int wave = tid >> 5, lane = tid & 31;

  for (int s = tid; s < S_TOT; s += 256) sms[s] = maxscore[(size_t)b * S_TOT + s];
  __syncthreads();

  for (int it = 0; it < NQ; ++it) {
    float best = NEG_BIG; int bi = 0x7fffffff;
    for (int s = tid; s < S_TOT; s += 256) {
      float v = sms[s];
      if (v > best) { best = v; bi = s; }   // increasing s keeps lowest index
    }
#pragma unroll
    for (int off = 16; off > 0; off >>= 1) {
      float vo = __shfl_down(best, off);
      int   io = __shfl_down(bi, off);
      if (vo > best || (vo == best && io < bi)) { best = vo; bi = io; }
    }
    if (lane == 0) { rv[wave] = best; ri[wave] = bi; }
    __syncthreads();
    if (tid == 0) {
      float B = rv[0]; int I = ri[0];
      for (int w = 1; w < 8; ++w)
        if (rv[w] > B || (rv[w] == B && ri[w] < I)) { B = rv[w]; I = ri[w]; }
      topk[b * NQ + it] = I;
      sms[I] = NEG_BIG;
    }
    __syncthreads();
  }
}

// ---------------------------------------------------------------------------
// Kernel 5: gather 16 selected tokens, rerun proj+LN (-> target), score
// (-> logits), and 3-layer box MLP with WMMA (-> ref points / sigmoid boxes).
// ---------------------------------------------------------------------------
#define P3_TILES 19    // ceil(300/16)

__global__ __launch_bounds__(256) void qs_phase3(
    const float* __restrict__ mm, const int* __restrict__ topk,
    const float* __restrict__ anchors,
    const bf16* __restrict__ projT, const bf16* __restrict__ scoreT,
    const bf16* __restrict__ b1T, const bf16* __restrict__ b2T,
    const bf16* __restrict__ b3T,
    const float* __restrict__ proj_b, const float* __restrict__ ln_g,
    const float* __restrict__ ln_b, const float* __restrict__ score_b,
    const float* __restrict__ box_b1, const float* __restrict__ box_b2,
    const float* __restrict__ box_b3,
    float* __restrict__ target, float* __restrict__ refp,
    float* __restrict__ boxes, float* __restrict__ logits) {
  __shared__ __align__(64) bf16 sA[16 * 256];   // A / encoded / h2
  __shared__ __align__(64) float sC[16 * 256];  // f32 GEMM results
  __shared__ __align__(64) bf16 sH[16 * 256];   // h1
  __shared__ int sIdx[16];
  __shared__ float sMu[16], sRstd[16];

  const int tid = threadIdx.x;
  const int wave = tid >> 5, lane = tid & 31;
  const int b  = blockIdx.x / P3_TILES;
  const int q0 = (blockIdx.x % P3_TILES) * 16;

  if (tid < 16) {
    int q = q0 + tid;
    sIdx[tid] = (q < NQ) ? topk[b * NQ + q] : 0;
  }
  __syncthreads();

  // gather masked_memory rows -> bf16 A
  for (int idx = tid; idx < 16 * 256; idx += 256) {
    int r = idx >> 8, c = idx & 255;
    float v = 0.f;
    if (q0 + r < NQ) v = mm[((size_t)b * S_TOT + sIdx[r]) * 256 + c];
    sA[idx] = (bf16)v;
  }
  __syncthreads();

  // proj: 16 N-tiles over 8 waves
  for (int t = wave; t < 16; t += 8) {
    v8f acc = {};
    acc = wmma_k256(sA, 256, projT, 256, 0, t * 16, lane, acc);
    int cb = t * 16 + (lane & 15);
    int rb = (lane >> 4) << 3;
    float bias = proj_b[cb];
#pragma unroll
    for (int i = 0; i < 8; ++i) sC[(rb + i) * 256 + cb] = acc[i] + bias;
  }
  __syncthreads();

  if (tid < 16) {
    float s1 = 0.f, s2 = 0.f;
    for (int c = 0; c < 256; ++c) { float v = sC[tid * 256 + c]; s1 += v; s2 += v * v; }
    float mu = s1 * (1.f / 256.f);
    float var = s2 * (1.f / 256.f) - mu * mu;
    sMu[tid] = mu; sRstd[tid] = rsqrtf(var + 1e-5f);
  }
  __syncthreads();

  // normalize -> target out + encoded bf16 (overwrites sA)
  for (int idx = tid; idx < 16 * 256; idx += 256) {
    int r = idx >> 8, c = idx & 255;
    float v = (sC[idx] - sMu[r]) * sRstd[r] * ln_g[c] + ln_b[c];
    int q = q0 + r;
    if (q < NQ) target[((size_t)b * NQ + q) * 256 + c] = v;
    sA[idx] = (bf16)v;
  }
  __syncthreads();

  // score head: 5 N-tiles, direct global store
  if (wave < 5) {
    v8f acc = {};
    acc = wmma_k256(sA, 256, scoreT, 80, 0, wave * 16, lane, acc);
    int cb = wave * 16 + (lane & 15);
    int rb = (lane >> 4) << 3;
    float bias = score_b[cb];
#pragma unroll
    for (int i = 0; i < 8; ++i) {
      int q = q0 + rb + i;
      if (q < NQ) logits[((size_t)b * NQ + q) * 80 + cb] = acc[i] + bias;
    }
  }

  // box layer 1: encoded -> relu -> sH (bf16)
  for (int t = wave; t < 16; t += 8) {
    v8f acc = {};
    acc = wmma_k256(sA, 256, b1T, 256, 0, t * 16, lane, acc);
    int cb = t * 16 + (lane & 15);
    int rb = (lane >> 4) << 3;
    float bias = box_b1[cb];
#pragma unroll
    for (int i = 0; i < 8; ++i)
      sH[(rb + i) * 256 + cb] = (bf16)fmaxf(acc[i] + bias, 0.f);
  }
  __syncthreads();

  // box layer 2: sH -> relu -> sA (bf16; encoded no longer needed)
  for (int t = wave; t < 16; t += 8) {
    v8f acc = {};
    acc = wmma_k256(sH, 256, b2T, 256, 0, t * 16, lane, acc);
    int cb = t * 16 + (lane & 15);
    int rb = (lane >> 4) << 3;
    float bias = box_b2[cb];
#pragma unroll
    for (int i = 0; i < 8; ++i)
      sA[(rb + i) * 256 + cb] = (bf16)fmaxf(acc[i] + bias, 0.f);
  }
  __syncthreads();

  // box layer 3: one padded 16-wide tile (cols 0..3 valid) + anchors
  if (wave == 0) {
    v8f acc = {};
    acc = wmma_k256(sA, 256, b3T, 16, 0, 0, lane, acc);
    int cb = lane & 15;
    int rb = (lane >> 4) << 3;
    if (cb < 4) {
      float bias = box_b3[cb];
#pragma unroll
      for (int i = 0; i < 8; ++i) {
        int r = rb + i;
        int q = q0 + r;
        if (q < NQ) {
          float u = acc[i] + bias + anchors[sIdx[r] * 4 + cb];
          size_t o = ((size_t)b * NQ + q) * 4 + cb;
          refp[o]  = u;
          boxes[o] = 1.f / (1.f + __expf(-u));
        }
      }
    }
  }
}

// ---------------------------------------------------------------------------
extern "C" void kernel_launch(void* const* d_in, const int* in_sizes, int n_in,
                              void* d_out, int out_size, void* d_ws, size_t ws_size,
                              hipStream_t stream) {
  const float* memory  = (const float*)d_in[0];
  // d_in[1] = spatial_shapes (static: [[80,80],[40,40],[20,20]]) -> unused
  const float* proj_w  = (const float*)d_in[2];
  const float* proj_b  = (const float*)d_in[3];
  const float* ln_g    = (const float*)d_in[4];
  const float* ln_b    = (const float*)d_in[5];
  const float* score_w = (const float*)d_in[6];
  const float* score_b = (const float*)d_in[7];
  const float* box_w1  = (const float*)d_in[8];
  const float* box_b1  = (const float*)d_in[9];
  const float* box_w2  = (const float*)d_in[10];
  const float* box_b2  = (const float*)d_in[11];
  const float* box_w3  = (const float*)d_in[12];
  const float* box_b3  = (const float*)d_in[13];

  char* ws = (char*)d_ws;
  size_t off = 0;
  auto take = [&](size_t n) {
    off = (off + 255) & ~(size_t)255;
    size_t o = off; off += n; return o;
  };
  float* anchors  = (float*)(ws + take((size_t)S_TOT * 4 * sizeof(float)));
  float* mask     = (float*)(ws + take((size_t)S_TOT * sizeof(float)));
  bf16*  projT    = (bf16*) (ws + take((size_t)256 * 256 * 2));
  bf16*  scoreT   = (bf16*) (ws + take((size_t)256 * 80 * 2));
  bf16*  b1T      = (bf16*) (ws + take((size_t)256 * 256 * 2));
  bf16*  b2T      = (bf16*) (ws + take((size_t)256 * 256 * 2));
  bf16*  b3T      = (bf16*) (ws + take((size_t)256 * 16 * 2));
  float* maxscore = (float*)(ws + take((size_t)BATCH * S_TOT * sizeof(float)));
  int*   topk     = (int*)  (ws + take((size_t)BATCH * NQ * sizeof(int)));

  float* out    = (float*)d_out;
  float* target = out;                                  // [32,300,256]
  float* refp   = target + (size_t)BATCH * NQ * 256;    // [32,300,4]
  float* boxes  = refp   + (size_t)BATCH * NQ * 4;      // [32,300,4]
  float* logits = boxes  + (size_t)BATCH * NQ * 4;      // [32,300,80]
  float* mm_out = logits + (size_t)BATCH * NQ * 80;     // [32,8400,256]

  qs_prep_weights<<<864, 256, 0, stream>>>(proj_w, score_w, box_w1, box_w2, box_w3,
                                           projT, scoreT, b1T, b2T, b3T);
  qs_anchors<<<(S_TOT + 255) / 256, 256, 0, stream>>>(anchors, mask);
  qs_phase2<<<BATCH * P2_TILES, 256, 0, stream>>>(memory, mask, projT, scoreT,
                                                  proj_b, ln_g, ln_b, score_b,
                                                  mm_out, maxscore);
  qs_topk<<<BATCH, 256, 0, stream>>>(maxscore, topk);
  qs_phase3<<<BATCH * P3_TILES, 256, 0, stream>>>(mm_out, topk, anchors,
                                                  projT, scoreT, b1T, b2T, b3T,
                                                  proj_b, ln_g, ln_b, score_b,
                                                  box_b1, box_b2, box_b3,
                                                  target, refp, boxes, logits);
}
